// CausalSelfAttention_23897198035162
// MI455X (gfx1250) — matrix operand order repaired and checked by reading, not compiled, not run on hardware
//
#include <hip/hip_runtime.h>
#include <hip/hip_bf16.h>
#include <stdint.h>

// CDNA5 / gfx1250, wave32. Mixed-precision attention: bf16 WMMA, f32 accum.
typedef __attribute__((ext_vector_type(16))) __bf16 v16bf;
typedef __attribute__((ext_vector_type(8)))  float  v8f;

// round-to-nearest-even f32 -> bf16
__device__ __forceinline__ unsigned short f2bf(float f) {
  union { float f; unsigned u; } v; v.f = f;
  unsigned r = v.u + 0x7fffu + ((v.u >> 16) & 1u);
  return (unsigned short)(r >> 16);
}

union Frag { uint4 q[2]; v16bf v; };

__device__ __forceinline__ Frag load_frag(const unsigned short* p, int o0, int o1) {
  Frag f;
  f.q[0] = *(const uint4*)(p + o0);   // 16B -> global_load_b128 / ds_load_b128
  f.q[1] = *(const uint4*)(p + o1);
  return f;
}

__device__ __forceinline__ v8f wmma_bf16(Frag a, Frag b, v8f c) {
  return __builtin_amdgcn_wmma_f32_16x16x32_bf16(
      /*neg_a=*/false, a.v, /*neg_b=*/false, b.v,
      /*c_mod=*/(short)0, c, /*reuse_a=*/false, /*reuse_b=*/false);
}

// ---------------- f32 -> bf16 conversion ----------------
__global__ void cvt_bf16(const float* __restrict__ src,
                         unsigned short* __restrict__ dst, int n) {
  int i = blockIdx.x * blockDim.x + threadIdx.x;
  if (i < n) dst[i] = f2bf(src[i]);
}

// ---------------- fused QKV projection ----------------
// y = x @ W^T + b ; X:[8192,1024]bf16, W:[3][1024][1024]bf16 (rows = out feat)
// Per-wave tile: 32 rows x 64 cols (one head-width) -> 2 A frags x 4 B frags,
// 8 WMMAs per K=32 step; ~2.7x less L2 traffic than 16x16 tiling.
// Q,K -> [B=4, H=16, T=2048, 64] bf16 ; V -> transposed [4,16,64,2048] bf16
__global__ void qkv_gemm(const unsigned short* __restrict__ X,
                         const unsigned short* __restrict__ W,
                         const float* __restrict__ bq,
                         const float* __restrict__ bk,
                         const float* __restrict__ bv,
                         unsigned short* __restrict__ Qo,
                         unsigned short* __restrict__ Ko,
                         unsigned short* __restrict__ Vt) {
  const int wave = threadIdx.x >> 5;
  int tile = blockIdx.x * 4 + wave;          // 3 * 256 * 16 = 12288 tiles
  int proj = tile / (256 * 16);
  int rem  = tile - proj * (256 * 16);
  int m0 = (rem >> 4) * 32;                  // 256 M-tiles of 32
  int nt = rem & 15;                         // 16 N-tiles of 64 (== head id)
  int n0 = nt * 64;
  int lane = threadIdx.x & 31;
  int n16 = lane & 15, half = lane >> 4;

  const unsigned short* A0 = X + (size_t)(m0 + n16) * 1024;
  const unsigned short* A1 = X + (size_t)(m0 + 16 + n16) * 1024;
  const unsigned short* Wp = W + (size_t)proj * 1024 * 1024;
  const unsigned short* B0 = Wp + (size_t)(n0 + n16) * 1024;
  const unsigned short* B1 = Wp + (size_t)(n0 + 16 + n16) * 1024;
  const unsigned short* B2 = Wp + (size_t)(n0 + 32 + n16) * 1024;
  const unsigned short* B3 = Wp + (size_t)(n0 + 48 + n16) * 1024;

  v8f acc[2][4];
#pragma unroll
  for (int i = 0; i < 2; ++i)
#pragma unroll
    for (int j = 0; j < 4; ++j) acc[i][j] = (v8f){};

  for (int k0 = 0; k0 < 1024; k0 += 32) {
    if (k0 + 256 < 1024) {                    // gfx1250 global_prefetch_b8
      __builtin_prefetch(A0 + k0 + 256, 0, 1);
      __builtin_prefetch(B0 + k0 + 256, 0, 1);
    }
    Frag a0 = load_frag(A0, k0 + 16 * half, k0 + 16 * half + 8);
    Frag a1 = load_frag(A1, k0 + 16 * half, k0 + 16 * half + 8);
    Frag b0 = load_frag(B0, k0 + 16 * half, k0 + 16 * half + 8);
    Frag b1 = load_frag(B1, k0 + 16 * half, k0 + 16 * half + 8);
    Frag b2 = load_frag(B2, k0 + 16 * half, k0 + 16 * half + 8);
    Frag b3 = load_frag(B3, k0 + 16 * half, k0 + 16 * half + 8);
    acc[0][0] = wmma_bf16(a0, b0, acc[0][0]);
    acc[1][0] = wmma_bf16(a1, b0, acc[1][0]);
    acc[0][1] = wmma_bf16(a0, b1, acc[0][1]);
    acc[1][1] = wmma_bf16(a1, b1, acc[1][1]);
    acc[0][2] = wmma_bf16(a0, b2, acc[0][2]);
    acc[1][2] = wmma_bf16(a1, b2, acc[1][2]);
    acc[0][3] = wmma_bf16(a0, b3, acc[0][3]);
    acc[1][3] = wmma_bf16(a1, b3, acc[1][3]);
  }

  const float* bias = (proj == 0) ? bq : (proj == 1) ? bk : bv;
  float bb[4];
#pragma unroll
  for (int j = 0; j < 4; ++j) bb[j] = bias[n0 + 16 * j + n16];
  int h = nt;                                 // tile spans exactly one head

  if (proj == 2) {                            // V stored transposed [4,16,64,2048]
#pragma unroll
    for (int j = 0; j < 4; ++j) {
      int d = 16 * j + n16;
#pragma unroll
      for (int i = 0; i < 2; ++i)
#pragma unroll
        for (int r = 0; r < 8; ++r) {
          int m = m0 + 16 * i + r + 8 * half;   // C/D layout row
          int bat = m >> 11, t = m & 2047;
          Vt[(((size_t)(bat * 16 + h)) * 64 + d) * 2048 + t] = f2bf(acc[i][j][r] + bb[j]);
        }
    }
  } else {
    unsigned short* dst = (proj == 0) ? Qo : Ko;   // [4,16,2048,64]
#pragma unroll
    for (int j = 0; j < 4; ++j) {
      int d = 16 * j + n16;
#pragma unroll
      for (int i = 0; i < 2; ++i)
#pragma unroll
        for (int r = 0; r < 8; ++r) {
          int m = m0 + 16 * i + r + 8 * half;
          int bat = m >> 11, t = m & 2047;
          dst[(((size_t)(bat * 16 + h)) * 2048 + t) * 64 + d] = f2bf(acc[i][j][r] + bb[j]);
        }
    }
  }
}

// ---------------- flash attention, causal ----------------
// One wave handles one (b,h, 16-row query tile). s processed in chunks of 32.
__global__ void attn(const unsigned short* __restrict__ Q,
                     const unsigned short* __restrict__ K,
                     const unsigned short* __restrict__ Vt,
                     unsigned short* __restrict__ Y) {   // [8192,1024] bf16
  __shared__ __align__(16) unsigned short pls[4][16 * 32];
  const int wave = threadIdx.x >> 5;
  int tile = blockIdx.x * 4 + wave;    // 4*16*128 = 8192 tiles
  int tq = tile & 127;                 // query tile in T
  int bh = tile >> 7;                  // b*16 + h
  int t0 = tq * 16;
  int lane = threadIdx.x & 31;
  int n16 = lane & 15, half = lane >> 4;

  const unsigned short* Qb = Q  + (size_t)bh * 2048 * 64;
  const unsigned short* Kb = K  + (size_t)bh * 2048 * 64;
  const unsigned short* Vb = Vt + (size_t)bh * 64 * 2048;
  unsigned short* lds = pls[wave];

  // Q tile as A-operand, hd=64 -> two K=32 fragments
  const unsigned short* qrow = Qb + (size_t)(t0 + n16) * 64;
  Frag q0 = load_frag(qrow, 16 * half,      16 * half + 8);
  Frag q1 = load_frag(qrow, 32 + 16 * half, 40 + 16 * half);

  v8f o0 = {}, o1 = {}, o2 = {}, o3 = {};
  float mrow[8], lrow[8];
#pragma unroll
  for (int r = 0; r < 8; ++r) { mrow[r] = -1e30f; lrow[r] = 0.f; }
  const float scale = 0.125f;          // 1/sqrt(64)

  for (int s0 = 0; s0 <= t0 + 15; s0 += 32) {
    // ---- S = Q K^T for 16x32 chunk (two 16-col WMMA tiles) ----
    const unsigned short* kr0 = Kb + (size_t)(s0 + n16) * 64;
    const unsigned short* kr1 = Kb + (size_t)(s0 + 16 + n16) * 64;
    Frag k00 = load_frag(kr0, 16 * half,      16 * half + 8);
    Frag k01 = load_frag(kr0, 32 + 16 * half, 40 + 16 * half);
    Frag k10 = load_frag(kr1, 16 * half,      16 * half + 8);
    Frag k11 = load_frag(kr1, 32 + 16 * half, 40 + 16 * half);
    v8f S0 = {}, S1 = {};
    S0 = wmma_bf16(q0, k00, S0); S0 = wmma_bf16(q1, k01, S0);
    S1 = wmma_bf16(q0, k10, S1); S1 = wmma_bf16(q1, k11, S1);

    // ---- scale + causal mask ----
#pragma unroll
    for (int r = 0; r < 8; ++r) {
      int t = t0 + r + 8 * half;
      S0[r] = (s0 + n16      <= t) ? S0[r] * scale : -1e30f;
      S1[r] = (s0 + 16 + n16 <= t) ? S1[r] * scale : -1e30f;
    }

    // ---- online softmax (row reductions inside 16-lane halves) ----
#pragma unroll
    for (int r = 0; r < 8; ++r) {
      float cmax = fmaxf(S0[r], S1[r]);
      cmax = fmaxf(cmax, __shfl_xor(cmax, 1, 32));
      cmax = fmaxf(cmax, __shfl_xor(cmax, 2, 32));
      cmax = fmaxf(cmax, __shfl_xor(cmax, 4, 32));
      cmax = fmaxf(cmax, __shfl_xor(cmax, 8, 32));
      float mnew  = fmaxf(mrow[r], cmax);
      float alpha = __expf(mrow[r] - mnew);
      float p0 = __expf(S0[r] - mnew);
      float p1 = __expf(S1[r] - mnew);
      float ps = p0 + p1;
      ps += __shfl_xor(ps, 1, 32);
      ps += __shfl_xor(ps, 2, 32);
      ps += __shfl_xor(ps, 4, 32);
      ps += __shfl_xor(ps, 8, 32);
      lrow[r] = lrow[r] * alpha + ps;
      mrow[r] = mnew;
      o0[r] *= alpha; o1[r] *= alpha; o2[r] *= alpha; o3[r] *= alpha;
      int mr = r + 8 * half;
      lds[mr * 32 + n16]      = f2bf(p0);   // stage P for layout change
      lds[mr * 32 + 16 + n16] = f2bf(p1);
    }
    asm volatile("s_wait_dscnt 0" ::: "memory");  // order LDS store -> load (same wave)

    // ---- re-read P as A-operand ----
    Frag p;
    const unsigned short* prow = lds + (size_t)n16 * 32;
    p.q[0] = *(const uint4*)(prow + 16 * half);
    p.q[1] = *(const uint4*)(prow + 16 * half + 8);

    // ---- O += P @ V ; V^T rows contiguous in s ----
    Frag v0 = load_frag(Vb + (size_t)(n16)      * 2048, s0 + 16 * half, s0 + 16 * half + 8);
    Frag v1 = load_frag(Vb + (size_t)(16 + n16) * 2048, s0 + 16 * half, s0 + 16 * half + 8);
    Frag v2 = load_frag(Vb + (size_t)(32 + n16) * 2048, s0 + 16 * half, s0 + 16 * half + 8);
    Frag v3 = load_frag(Vb + (size_t)(48 + n16) * 2048, s0 + 16 * half, s0 + 16 * half + 8);
    o0 = wmma_bf16(p, v0, o0);
    o1 = wmma_bf16(p, v1, o1);
    o2 = wmma_bf16(p, v2, o2);
    o3 = wmma_bf16(p, v3, o3);
  }

  // ---- normalize + store y (bf16, [B,T,C] layout) ----
  int b = bh >> 4, h = bh & 15;
#pragma unroll
  for (int r = 0; r < 8; ++r) {
    float inv = 1.0f / lrow[r];
    int t = t0 + r + 8 * half;
    size_t off = ((size_t)(b * 2048 + t)) * 1024 + h * 64;
    Y[off + n16]      = f2bf(o0[r] * inv);
    Y[off + 16 + n16] = f2bf(o1[r] * inv);
    Y[off + 32 + n16] = f2bf(o2[r] * inv);
    Y[off + 48 + n16] = f2bf(o3[r] * inv);
  }
}

// ---------------- output projection: out = Y @ Wp^T + bp (f32 out) ----------------
// Same 32x64 per-wave tiling as qkv_gemm.
__global__ void out_gemm(const unsigned short* __restrict__ Yb,
                         const unsigned short* __restrict__ Wp,
                         const float* __restrict__ bp,
                         float* __restrict__ out) {
  const int wave = threadIdx.x >> 5;
  int tile = blockIdx.x * 4 + wave;          // 256 * 16 = 4096 tiles
  int m0 = (tile >> 4) * 32;
  int n0 = (tile & 15) * 64;
  int lane = threadIdx.x & 31;
  int n16 = lane & 15, half = lane >> 4;

  const unsigned short* A0 = Yb + (size_t)(m0 + n16) * 1024;
  const unsigned short* A1 = Yb + (size_t)(m0 + 16 + n16) * 1024;
  const unsigned short* B0 = Wp + (size_t)(n0 + n16) * 1024;
  const unsigned short* B1 = Wp + (size_t)(n0 + 16 + n16) * 1024;
  const unsigned short* B2 = Wp + (size_t)(n0 + 32 + n16) * 1024;
  const unsigned short* B3 = Wp + (size_t)(n0 + 48 + n16) * 1024;

  v8f acc[2][4];
#pragma unroll
  for (int i = 0; i < 2; ++i)
#pragma unroll
    for (int j = 0; j < 4; ++j) acc[i][j] = (v8f){};

  for (int k0 = 0; k0 < 1024; k0 += 32) {
    if (k0 + 256 < 1024) {
      __builtin_prefetch(A0 + k0 + 256, 0, 1);
      __builtin_prefetch(B0 + k0 + 256, 0, 1);
    }
    Frag a0 = load_frag(A0, k0 + 16 * half, k0 + 16 * half + 8);
    Frag a1 = load_frag(A1, k0 + 16 * half, k0 + 16 * half + 8);
    Frag b0 = load_frag(B0, k0 + 16 * half, k0 + 16 * half + 8);
    Frag b1 = load_frag(B1, k0 + 16 * half, k0 + 16 * half + 8);
    Frag b2 = load_frag(B2, k0 + 16 * half, k0 + 16 * half + 8);
    Frag b3 = load_frag(B3, k0 + 16 * half, k0 + 16 * half + 8);
    acc[0][0] = wmma_bf16(a0, b0, acc[0][0]);
    acc[1][0] = wmma_bf16(a1, b0, acc[1][0]);
    acc[0][1] = wmma_bf16(a0, b1, acc[0][1]);
    acc[1][1] = wmma_bf16(a1, b1, acc[1][1]);
    acc[0][2] = wmma_bf16(a0, b2, acc[0][2]);
    acc[1][2] = wmma_bf16(a1, b2, acc[1][2]);
    acc[0][3] = wmma_bf16(a0, b3, acc[0][3]);
    acc[1][3] = wmma_bf16(a1, b3, acc[1][3]);
  }

  float bb[4];
#pragma unroll
  for (int j = 0; j < 4; ++j) bb[j] = bp[n0 + 16 * j + n16];
#pragma unroll
  for (int j = 0; j < 4; ++j)
#pragma unroll
    for (int i = 0; i < 2; ++i)
#pragma unroll
      for (int r = 0; r < 8; ++r) {
        int m = m0 + 16 * i + r + 8 * half;
        out[(size_t)m * 1024 + n0 + 16 * j + n16] = acc[i][j][r] + bb[j];
      }
}

extern "C" void kernel_launch(void* const* d_in, const int* in_sizes, int n_in,
                              void* d_out, int out_size, void* d_ws, size_t ws_size,
                              hipStream_t stream) {
  (void)in_sizes; (void)n_in; (void)out_size; (void)ws_size;
  const float* query = (const float*)d_in[0];
  const float* Wq = (const float*)d_in[1];
  const float* bq = (const float*)d_in[2];
  const float* Wk = (const float*)d_in[3];
  const float* bk = (const float*)d_in[4];
  const float* Wv = (const float*)d_in[5];
  const float* bv = (const float*)d_in[6];
  const float* Wp = (const float*)d_in[7];
  const float* bp = (const float*)d_in[8];
  float* out = (float*)d_out;
  unsigned short* ws = (unsigned short*)d_ws;

  const size_t MC = 8192ull * 1024;   // 8,388,608 elements
  const size_t CC = 1024ull * 1024;   // 1,048,576 elements
  // workspace layout (bf16 elements), total ~88 MiB:
  unsigned short* Xb  = ws;            // [8192,1024]
  unsigned short* Wb  = Xb  + MC;      // [3][1024][1024] (q,k,v)
  unsigned short* Wpb = Wb  + 3 * CC;  // [1024,1024]
  unsigned short* Qo  = Wpb + CC;      // [4,16,2048,64]
  unsigned short* Ko  = Qo  + MC;      // [4,16,2048,64]
  unsigned short* Vt  = Ko  + MC;      // [4,16,64,2048]
  unsigned short* Yb  = Vt  + MC;      // [8192,1024]

  const int thr = 256;
  cvt_bf16<<<(int)((MC + thr - 1) / thr), thr, 0, stream>>>(query, Xb, (int)MC);
  cvt_bf16<<<(int)((CC + thr - 1) / thr), thr, 0, stream>>>(Wq, Wb + 0 * CC, (int)CC);
  cvt_bf16<<<(int)((CC + thr - 1) / thr), thr, 0, stream>>>(Wk, Wb + 1 * CC, (int)CC);
  cvt_bf16<<<(int)((CC + thr - 1) / thr), thr, 0, stream>>>(Wv, Wb + 2 * CC, (int)CC);
  cvt_bf16<<<(int)((CC + thr - 1) / thr), thr, 0, stream>>>(Wp, Wpb, (int)CC);

  qkv_gemm<<<(3 * 256 * 16) / 4, 128, 0, stream>>>(Xb, Wb, bq, bk, bv, Qo, Ko, Vt);
  attn<<<8192 / 4, 128, 0, stream>>>(Qo, Ko, Vt, Yb);
  out_gemm<<<(256 * 16) / 4, 128, 0, stream>>>(Yb, Wpb, bp, out);
}
